// Mp_encoder_41437844471878
// MI455X (gfx1250) — matrix operand order repaired and checked by reading, not compiled, not run on hardware
//
#include <hip/hip_runtime.h>

#define N_NODES 4096
#define DIM 256
#define NPATH 3
#define MAXDEG 17
#define TAU 0.8f
#define LAM 0.5f
#define BETA_W 1.0f
#define MARGIN 0.1f
#define EPS_PD 1e-6f

typedef __bf16 bf16;
typedef __attribute__((ext_vector_type(16))) __bf16 v16bf;
typedef __attribute__((ext_vector_type(8)))  __bf16 v8bf;
typedef __attribute__((ext_vector_type(8)))  float  v8f;

typedef int v4i_ __attribute__((vector_size(16)));
typedef __attribute__((address_space(1))) v4i_ as1_v4i;
typedef __attribute__((address_space(3))) v4i_ as3_v4i;

#if __has_builtin(__builtin_amdgcn_global_load_async_to_lds_b128) && \
    __has_builtin(__builtin_amdgcn_s_wait_asynccnt)
#define USE_ASYNC_LDS 1
#else
#define USE_ASYNC_LDS 0
#endif

// ---------------------------------------------------------------- utilities

__device__ __forceinline__ float wave_red32(float v) {
#pragma unroll
  for (int m = 16; m >= 1; m >>= 1) v += __shfl_xor(v, m, 32);
  return v;
}

// A fragment (16x32, MxK), row-major A with leading dim lda.
// Lane L<16: row L, K in {0..7, 16..23}; lane L>=16: row L-16, K in {8..15, 24..31}.
__device__ __forceinline__ v16bf load_frag_a(const bf16* __restrict__ A, int lda, int lane) {
  const int row = lane & 15;
  const int kb  = (lane >> 4) << 3;             // 0 or 8
  const bf16* p = A + (size_t)row * lda + kb;
  v8bf lo = *(const v8bf*)(p);
  v8bf hi = *(const v8bf*)(p + 16);
  v16bf r;
#pragma unroll
  for (int i = 0; i < 8; i++) { r[i] = lo[i]; r[i + 8] = hi[i]; }
  return r;
}

// B fragment (32x16, KxN) where B = Bt^T, Bt row-major NxK with leading dim ldb.
// Lane L<16: col L, K in {0..7, 16..23}; lane L>=16: col L-16, K in {8..15, 24..31}.
__device__ __forceinline__ v16bf load_frag_bt(const bf16* __restrict__ Bt, int ldb, int lane) {
  const int col = lane & 15;
  const int kb  = (lane >> 4) << 3;             // 0 or 8
  const bf16* p = Bt + (size_t)col * ldb + kb;
  v8bf lo = *(const v8bf*)(p);
  v8bf hi = *(const v8bf*)(p + 16);
  v16bf r;
#pragma unroll
  for (int i = 0; i < 8; i++) { r[i] = lo[i]; r[i + 8] = hi[i]; }
  return r;
}

// ---------------------------------------------------------------- small kernels

__global__ void k_zero(float* __restrict__ p, int n) {
  int i = blockIdx.x * blockDim.x + threadIdx.x;
  if (i < n) p[i] = 0.f;
}

__global__ void k_cvt(const float* __restrict__ s, bf16* __restrict__ d, int n) {
  int i = blockIdx.x * blockDim.x + threadIdx.x;
  if (i < n) d[i] = (bf16)s[i];
}

// ---------------------------------------------------------------- generic C = act(A @ Bt^T + bias)
// A: MxK bf16 row-major, Bt: NxK bf16 row-major. One wave per 16x16 tile.
// act: 0 = none, 2 = ELU, 3 = tanh. Either Cf (f32) or Cb (bf16) may be null.
__global__ __launch_bounds__(32) void k_gemm_xt(const bf16* __restrict__ A,
                                                const bf16* __restrict__ Bt,
                                                const float* __restrict__ bias,
                                                float* __restrict__ Cf,
                                                bf16* __restrict__ Cb,
                                                int ncols, int K, int act) {
  const int lane = threadIdx.x;
  const int n0 = blockIdx.x << 4;
  const int m0 = blockIdx.y << 4;
  v8f acc = {};
  for (int k0 = 0; k0 < K; k0 += 32) {
    v16bf a = load_frag_a(A + (size_t)m0 * K + k0, K, lane);
    v16bf b = load_frag_bt(Bt + (size_t)n0 * K + k0, K, lane);
    acc = __builtin_amdgcn_wmma_f32_16x16x32_bf16(false, a, false, b, (short)0, acc, false, false);
  }
  const int col = lane & 15;
  const int rb  = (lane >> 4) << 3;
  const float bv = bias ? bias[n0 + col] : 0.f;
#pragma unroll
  for (int r = 0; r < 8; r++) {
    float v = acc[r] + bv;
    if (act == 2)      v = v > 0.f ? v : expm1f(v);
    else if (act == 3) v = tanhf(v);
    size_t idx = (size_t)(m0 + rb + r) * ncols + n0 + col;
    if (Cf) Cf[idx] = v;
    if (Cb) Cb[idx] = (bf16)v;
  }
}

// ---------------------------------------------------------------- sparse adjacency extraction
// One wave per row: ballot-scan 4096 cols; nonzero columns in ascending order (== stable argsort),
// plus the first 17 zero columns for the negative set.
__global__ __launch_bounds__(32) void k_extract(const float* __restrict__ adj,
                                                int* __restrict__ nbr,
                                                int* __restrict__ negi,
                                                int* __restrict__ deg) {
  const int row = blockIdx.x;
  const float* a = adj + (size_t)row * N_NODES;
  const int lane = threadIdx.x;
  __shared__ int sidx[64];
  int base = 0;
  for (int c0 = 0; c0 < N_NODES; c0 += 32) {
    float v = a[c0 + lane];
    unsigned mask = __builtin_amdgcn_ballot_w32(v > 0.f);
    if (v > 0.f) {
      int pidx = base + __builtin_popcount(mask & ((1u << lane) - 1u));
      if (pidx < 64) sidx[pidx] = c0 + lane;
    }
    base += __builtin_popcount(mask);
  }
  __syncthreads();
  int d = base < MAXDEG ? base : MAXDEG;
  if (lane == 0) deg[row] = base;
  if (lane < MAXDEG) nbr[(size_t)row * MAXDEG + lane] = (lane < d) ? sidx[lane] : row;
  if (lane == 0) {
    int c = 0;
    for (int j = 0; j < 64 && c < MAXDEG; j++)
      if (!(a[j] > 0.f)) negi[(size_t)row * MAXDEG + (c++)] = j;
  }
}

// ---------------------------------------------------------------- GCN aggregation: emb = PReLU(adj @ XW + b)
__global__ void k_gather_gcn(const float* __restrict__ XW, const int* __restrict__ nbr,
                             const int* __restrict__ deg, const float* __restrict__ b,
                             const float* __restrict__ prelu_a, int path,
                             float* __restrict__ emb, bf16* __restrict__ emb_bf) {
  const int row = blockIdx.x, d = threadIdx.x;
  int dg = deg[row]; if (dg > MAXDEG) dg = MAXDEG;
  const int* nb = nbr + (size_t)row * MAXDEG;
  float acc = 0.f;
  for (int t = 0; t < dg; t++) acc += XW[(size_t)nb[t] * DIM + d];
  float v = acc / (float)dg + b[d];
  float al = prelu_a[path];
  v = v > 0.f ? v : al * v;
  emb[(size_t)row * DIM + d] = v;
  emb_bf[(size_t)row * DIM + d] = (bf16)v;
}

// ---------------------------------------------------------------- triplet node loss
__global__ void k_node_loss(const float* __restrict__ emb, const int* __restrict__ nbr,
                            const int* __restrict__ negi, const int* __restrict__ deg,
                            float* __restrict__ out_acc) {
  const int i = blockIdx.x;
  const int tid = threadIdx.x, lane = tid & 31, w = tid >> 5;
  __shared__ float s;
  if (tid == 0) s = 0.f;
  __syncthreads();
  int dg = deg[i]; if (dg > MAXDEG) dg = MAXDEG;
  float local = 0.f;
  for (int j = w; j < MAXDEG; j += 8) {
    if (j < dg) {
      int pj = nbr[(size_t)i * MAXDEG + j];
      int nj = negi[(size_t)i * MAXDEG + j];
      float dp = 0.f, dn = 0.f;
#pragma unroll
      for (int t = 0; t < 8; t++) {
        int d = lane * 8 + t;
        float a = emb[(size_t)i * DIM + d];
        float x = a - emb[(size_t)pj * DIM + d] + EPS_PD; dp += x * x;
        float y = a - emb[(size_t)nj * DIM + d] + EPS_PD; dn += y * y;
      }
      dp = wave_red32(dp); dn = wave_red32(dn);
      float per = dp - dn + MARGIN;
      if (lane == 0 && per > 0.f) local += per;
    }
  }
  if (lane == 0) atomicAdd(&s, local);
  __syncthreads();
  if (tid == 0) atomicAdd(out_acc, s / (float)dg);
}

// ---------------------------------------------------------------- row-normalize projection (f32 + bf16 outputs)
__global__ __launch_bounds__(32) void k_rownorm(const float* __restrict__ P,
                                                float* __restrict__ Pf, bf16* __restrict__ Pb,
                                                float extra) {
  const int row = blockIdx.x, lane = threadIdx.x;
  const float* p = P + (size_t)row * DIM + lane * 8;
  float v[8]; float ss = 0.f;
#pragma unroll
  for (int t = 0; t < 8; t++) { v[t] = p[t]; ss += v[t] * v[t]; }
  ss = wave_red32(ss);
  float sc = extra * rsqrtf(ss);
#pragma unroll
  for (int t = 0; t < 8; t++) {
    float o = v[t] * sc;
    size_t idx = (size_t)row * DIM + lane * 8 + t;
    Pf[idx] = o;
    Pb[idx] = (bf16)o;
  }
}

// ---------------------------------------------------------------- fused S = exp(P1s @ P2n^T): row/col sums only
// 8-wave block computes a 128x128 panel. B panel (128 x 256 bf16 = 64 KB) is staged once in LDS
// (async-to-LDS when the toolchain exposes it); each wave caches its 16x256 A strip in VGPRs.
__global__ __launch_bounds__(256) void k_sim_sums(const bf16* __restrict__ P1,
                                                  const bf16* __restrict__ P2,
                                                  float* __restrict__ rowsum,
                                                  float* __restrict__ colsum) {
  __shared__ bf16 Bs[128 * DIM];                       // 64 KB
  const int tid  = threadIdx.x;
  const int lane = tid & 31;
  const int wave = tid >> 5;
  const int m0   = (blockIdx.y << 7) + (wave << 4);    // this wave's 16-row strip
  const int np0  = blockIdx.x << 7;                    // 128-column panel base

  // ---- stage B panel: rows np0..np0+127 of P2 (512 B each) -> LDS, 16 B per thread per step.
  {
    const char* src = (const char*)(P2 + (size_t)np0 * DIM);
    char*       dst = (char*)Bs;
#pragma unroll
    for (int it = 0; it < 16; ++it) {
      int off = (it * 256 + tid) * 16;                 // 0 .. 65535
#if USE_ASYNC_LDS
      __builtin_amdgcn_global_load_async_to_lds_b128(
          (as1_v4i*)(uintptr_t)(src + off),
          (as3_v4i*)(uint32_t)(uintptr_t)(dst + off),
          0, 0);
#else
      *(float4*)(dst + off) = *(const float4*)(src + off);
#endif
    }
#if USE_ASYNC_LDS
    __builtin_amdgcn_s_wait_asynccnt(0);
#endif
    __syncthreads();
  }

  // ---- cache A strip (16 x 256) in registers.
  v16bf aA[8];
#pragma unroll
  for (int kk = 0; kk < 8; kk++)
    aA[kk] = load_frag_a(P1 + (size_t)m0 * DIM + kk * 32, DIM, lane);

  // ---- 8 n-tiles against the LDS panel.
#pragma unroll 1
  for (int nt = 0; nt < 8; nt++) {
    const int nl = nt << 4;                            // local col base in panel
    v8f acc = {};
#pragma unroll
    for (int kk = 0; kk < 8; kk++) {
      v16bf b = load_frag_bt(Bs + (size_t)nl * DIM + kk * 32, DIM, lane);
      acc = __builtin_amdgcn_wmma_f32_16x16x32_bf16(false, aA[kk], false, b, (short)0, acc, false, false);
    }
#pragma unroll
    for (int r = 0; r < 8; r++) acc[r] = __expf(acc[r]);
    // column sums: lane holds column np0+nl+(lane&15); rows split between half-waves.
    float colv = 0.f;
#pragma unroll
    for (int r = 0; r < 8; r++) colv += acc[r];
    colv += __shfl_xor(colv, 16, 32);
    if (lane < 16) atomicAdd(&colsum[np0 + nl + lane], colv);
    // row sums: reduce across the 16 lanes of each half-wave.
#pragma unroll
    for (int r = 0; r < 8; r++) {
      float v = acc[r];
#pragma unroll
      for (int m = 8; m >= 1; m >>= 1) v += __shfl_xor(v, m, 32);
      if ((lane & 15) == 0) atomicAdd(&rowsum[m0 + ((lane >> 4) << 3) + r], v);
    }
  }
}

// ---------------------------------------------------------------- positive-masked sums over <=17 neighbors
__global__ __launch_bounds__(32) void k_pos_sums(const float* __restrict__ P1s,
                                                 const float* __restrict__ P2n,
                                                 const int* __restrict__ nbr,
                                                 const int* __restrict__ deg,
                                                 float* __restrict__ posrow,
                                                 float* __restrict__ poscol) {
  const int i = blockIdx.x, lane = threadIdx.x;
  float a1[8], a2[8];
#pragma unroll
  for (int t = 0; t < 8; t++) {
    a1[t] = P1s[(size_t)i * DIM + lane * 8 + t];
    a2[t] = P2n[(size_t)i * DIM + lane * 8 + t];
  }
  int dg = deg[i]; if (dg > MAXDEG) dg = MAXDEG;
  float sr = 0.f, sc = 0.f;
  for (int t = 0; t < dg; t++) {
    int j = nbr[(size_t)i * MAXDEG + t];
    float d12 = 0.f, d21 = 0.f;
#pragma unroll
    for (int u = 0; u < 8; u++) {
      int d = lane * 8 + u;
      d12 += a1[u] * P2n[(size_t)j * DIM + d];
      d21 += P1s[(size_t)j * DIM + d] * a2[u];
    }
    d12 = wave_red32(d12); d21 = wave_red32(d21);
    sr += __expf(d12); sc += __expf(d21);
  }
  if (lane == 0) { posrow[i] = sr / (float)dg; poscol[i] = sc / (float)dg; }
}

__global__ void k_cfin(const float* __restrict__ rs, const float* __restrict__ cs,
                       const float* __restrict__ pr, const float* __restrict__ pc,
                       float* __restrict__ closs) {
  int i = blockIdx.x * blockDim.x + threadIdx.x;
  float l12 = -logf(pr[i] / (rs[i] + 1e-8f));
  float l21 = -logf(pc[i] / (cs[i] + 1e-8f));
  atomicAdd(closs, (LAM * l12 + (1.f - LAM) * l21) * (1.f / N_NODES));
}

// ---------------------------------------------------------------- attention
__global__ void k_colmean(const float* __restrict__ T, float* __restrict__ sp) {
  const int d = blockIdx.x, tid = threadIdx.x;
  __shared__ float red[256];
  float s = 0.f;
  for (int r = tid; r < N_NODES; r += 256) s += T[(size_t)r * DIM + d];
  red[tid] = s; __syncthreads();
  for (int st = 128; st > 0; st >>= 1) { if (tid < st) red[tid] += red[tid + st]; __syncthreads(); }
  if (tid == 0) sp[d] = red[0] / (float)N_NODES;
}

__global__ void k_att(const float* __restrict__ sp, const float* __restrict__ att_vec,
                      float* __restrict__ beta) {
  __shared__ float red[256];
  __shared__ float w[NPATH];
  int tid = threadIdx.x;
  float av = att_vec[tid];
  for (int p = 0; p < NPATH; p++) {
    red[tid] = sp[p * DIM + tid] * av; __syncthreads();
    for (int st = 128; st > 0; st >>= 1) { if (tid < st) red[tid] += red[tid + st]; __syncthreads(); }
    if (tid == 0) w[p] = red[0];
    __syncthreads();
  }
  if (tid == 0) {
    float mx = fmaxf(w[0], fmaxf(w[1], w[2]));
    float e0 = __expf(w[0] - mx), e1 = __expf(w[1] - mx), e2 = __expf(w[2] - mx);
    float s = e0 + e1 + e2;
    beta[0] = e0 / s; beta[1] = e1 / s; beta[2] = e2 / s;
  }
}

__global__ void k_zmix(const float* __restrict__ emb, const float* __restrict__ beta,
                       float* __restrict__ out) {
  size_t i = (size_t)blockIdx.x * 256 + threadIdx.x;
  const size_t ND = (size_t)N_NODES * DIM;
  out[i] = beta[0] * emb[i] + beta[1] * emb[ND + i] + beta[2] * emb[2 * ND + i];
}

__global__ void k_final(const float* __restrict__ acc, float* __restrict__ out) {
  out[0] = acc[0] + BETA_W * acc[1];
}

// ---------------------------------------------------------------- launcher

extern "C" void kernel_launch(void* const* d_in, const int* in_sizes, int n_in,
                              void* d_out, int out_size, void* d_ws, size_t ws_size,
                              hipStream_t stream) {
  (void)in_sizes; (void)n_in; (void)out_size; (void)ws_size;
  const float* h        = (const float*)d_in[0];
  const float* mps      = (const float*)d_in[1];
  const float* W_gcn    = (const float*)d_in[2];
  const float* b_gcn    = (const float*)d_in[3];
  const float* prelu_a  = (const float*)d_in[4];
  const float* att_fc_W = (const float*)d_in[5];
  const float* att_fc_b = (const float*)d_in[6];
  const float* att_vec  = (const float*)d_in[7];
  const float* proj_W1  = (const float*)d_in[8];
  const float* proj_b1  = (const float*)d_in[9];
  const float* proj_W2  = (const float*)d_in[10];
  const float* proj_b2  = (const float*)d_in[11];
  float* out = (float*)d_out;

  char* ws = (char*)d_ws;
  size_t off = 0;
  auto alloc = [&](size_t bytes) -> char* {
    char* r = ws + off;
    off += (bytes + 255) & ~(size_t)255;
    return r;
  };
  const size_t ND = (size_t)N_NODES * DIM;
  bf16* h_bf    = (bf16*)alloc(ND * 2);
  bf16* Wg_bf   = (bf16*)alloc((size_t)NPATH * DIM * DIM * 2);
  bf16* attW_bf = (bf16*)alloc((size_t)DIM * DIM * 2);
  bf16* W1_bf   = (bf16*)alloc((size_t)DIM * DIM * 2);
  bf16* W2_bf   = (bf16*)alloc((size_t)DIM * DIM * 2);
  float* XW     = (float*)alloc(ND * 4);
  int*   nbr    = (int*)alloc((size_t)NPATH * N_NODES * MAXDEG * 4);
  int*   negi   = (int*)alloc((size_t)NPATH * N_NODES * MAXDEG * 4);
  int*   deg    = (int*)alloc((size_t)NPATH * N_NODES * 4);
  float* emb    = (float*)alloc((size_t)NPATH * ND * 4);
  bf16*  emb_bf = (bf16*)alloc((size_t)NPATH * ND * 2);
  bf16*  t_bf   = (bf16*)alloc(ND * 2);
  float* t_f    = (float*)alloc(ND * 4);
  float* p1     = (float*)alloc(ND * 4);
  float* p2     = (float*)alloc(ND * 4);
  float* p1s_f  = (float*)alloc(ND * 4);
  float* p2n_f  = (float*)alloc(ND * 4);
  bf16*  p1s_bf = (bf16*)alloc(ND * 2);
  bf16*  p2n_bf = (bf16*)alloc(ND * 2);
  float* sums   = (float*)alloc((size_t)4 * N_NODES * 4); // rowsum,colsum,posrow,poscol
  float* sp     = (float*)alloc((size_t)NPATH * DIM * 4);
  float* beta   = (float*)alloc(16);
  float* acc    = (float*)alloc(16); // [0]=node loss, [1]=contrast loss

  k_zero<<<1, 32, 0, stream>>>(acc, 2);
  k_cvt<<<((int)ND + 255) / 256, 256, 0, stream>>>(h, h_bf, (int)ND);
  k_cvt<<<(NPATH * DIM * DIM + 255) / 256, 256, 0, stream>>>(W_gcn, Wg_bf, NPATH * DIM * DIM);
  k_cvt<<<(DIM * DIM + 255) / 256, 256, 0, stream>>>(att_fc_W, attW_bf, DIM * DIM);
  k_cvt<<<(DIM * DIM + 255) / 256, 256, 0, stream>>>(proj_W1, W1_bf, DIM * DIM);
  k_cvt<<<(DIM * DIM + 255) / 256, 256, 0, stream>>>(proj_W2, W2_bf, DIM * DIM);

  dim3 gemm_grid(DIM / 16, N_NODES / 16);
  for (int p = 0; p < NPATH; p++) {
    k_gemm_xt<<<gemm_grid, 32, 0, stream>>>(h_bf, Wg_bf + (size_t)p * DIM * DIM,
                                            nullptr, XW, nullptr, DIM, DIM, 0);
    k_extract<<<N_NODES, 32, 0, stream>>>(mps + (size_t)p * N_NODES * N_NODES,
                                          nbr + (size_t)p * N_NODES * MAXDEG,
                                          negi + (size_t)p * N_NODES * MAXDEG,
                                          deg + (size_t)p * N_NODES);
    k_gather_gcn<<<N_NODES, DIM, 0, stream>>>(XW, nbr + (size_t)p * N_NODES * MAXDEG,
                                              deg + (size_t)p * N_NODES,
                                              b_gcn + (size_t)p * DIM, prelu_a, p,
                                              emb + (size_t)p * ND, emb_bf + (size_t)p * ND);
    k_node_loss<<<N_NODES, 256, 0, stream>>>(emb + (size_t)p * ND,
                                             nbr + (size_t)p * N_NODES * MAXDEG,
                                             negi + (size_t)p * N_NODES * MAXDEG,
                                             deg + (size_t)p * N_NODES, acc);
  }

  const int pair_a[2] = {0, 1};
  for (int q = 0; q < 2; q++) {
    int a = pair_a[q], c = NPATH - 1;
    k_gemm_xt<<<gemm_grid, 32, 0, stream>>>(emb_bf + (size_t)a * ND, W1_bf, proj_b1,
                                            nullptr, t_bf, DIM, DIM, 2);
    k_gemm_xt<<<gemm_grid, 32, 0, stream>>>(t_bf, W2_bf, proj_b2, p1, nullptr, DIM, DIM, 0);
    k_gemm_xt<<<gemm_grid, 32, 0, stream>>>(emb_bf + (size_t)c * ND, W1_bf, proj_b1,
                                            nullptr, t_bf, DIM, DIM, 2);
    k_gemm_xt<<<gemm_grid, 32, 0, stream>>>(t_bf, W2_bf, proj_b2, p2, nullptr, DIM, DIM, 0);
    k_rownorm<<<N_NODES, 32, 0, stream>>>(p1, p1s_f, p1s_bf, 1.0f / TAU);
    k_rownorm<<<N_NODES, 32, 0, stream>>>(p2, p2n_f, p2n_bf, 1.0f);
    k_zero<<<(4 * N_NODES + 255) / 256, 256, 0, stream>>>(sums, 4 * N_NODES);
    dim3 sgrid(N_NODES / 128, N_NODES / 128);
    k_sim_sums<<<sgrid, 256, 0, stream>>>(p1s_bf, p2n_bf, sums, sums + N_NODES);
    k_pos_sums<<<N_NODES, 32, 0, stream>>>(p1s_f, p2n_f,
                                           nbr + (size_t)a * N_NODES * MAXDEG,
                                           deg + (size_t)a * N_NODES,
                                           sums + 2 * N_NODES, sums + 3 * N_NODES);
    k_cfin<<<N_NODES / 256, 256, 0, stream>>>(sums, sums + N_NODES,
                                              sums + 2 * N_NODES, sums + 3 * N_NODES, acc + 1);
  }

  for (int p = 0; p < NPATH; p++) {
    k_gemm_xt<<<gemm_grid, 32, 0, stream>>>(emb_bf + (size_t)p * ND, attW_bf, att_fc_b,
                                            t_f, nullptr, DIM, DIM, 3);
    k_colmean<<<DIM, 256, 0, stream>>>(t_f, sp + (size_t)p * DIM);
  }
  k_att<<<1, DIM, 0, stream>>>(sp, att_vec, beta);
  k_zmix<<<(int)(ND / 256), 256, 0, stream>>>(emb, beta, out);
  k_final<<<1, 1, 0, stream>>>(acc, out + ND);
}
